// Model_26560077758479
// MI455X (gfx1250) — compile-verified, matrix operand order checked by reading, not run on hardware
//
#include <hip/hip_runtime.h>
#include <hip/hip_bf16.h>
#include <math.h>

// Model dims (fixed by reference)
#define BB 16
#define TT 2048
#define DD 128
#define HH 256

typedef __attribute__((ext_vector_type(16))) _Float16 v16h;
typedef __attribute__((ext_vector_type(8)))  _Float16 v8h;
typedef __attribute__((ext_vector_type(8)))  float    v8f;
typedef __attribute__((ext_vector_type(4)))  int      v4i;

// Async global->LDS path (gfx1250); guarded so either toolchain compiles.
#if defined(__HIP_DEVICE_COMPILE__) && defined(__gfx1250__)
#if __has_builtin(__builtin_amdgcn_global_load_async_to_lds_b128) && \
    __has_builtin(__builtin_amdgcn_s_wait_asynccnt)
#define HAVE_ASYNC_LDS 1
#endif
#endif

union F16Frag { v16h v; v8h g[2]; _Float16 h[16]; };

// WMMA f16 fragment layout (16x32 A / 32x16 B): halves 0..7 and 8..15 are
// K-contiguous runs of 8 -> load each fragment as two 16-byte vectors.
//   lanes 0-15 : halves 0..7 -> K kb+0..7,  halves 8..15 -> K kb+16..23
//   lanes 16-31: halves 0..7 -> K kb+8..15, halves 8..15 -> K kb+24..31
__device__ __forceinline__ v16h load_frag16(const _Float16* __restrict__ row,
                                            int kb, int lane) {
  F16Frag f;
  const int base = kb + ((lane & 16) ? 8 : 0);
  f.g[0] = *(const v8h*)(row + base);
  f.g[1] = *(const v8h*)(row + base + 16);
  return f.v;
}

// ---------------------------------------------------------------------------
// f32 -> f16 conversion (activations and one-time weight conversion)
// ---------------------------------------------------------------------------
__global__ void cvt_f32_f16(const float* __restrict__ src,
                            _Float16* __restrict__ dst, int n) {
  int i = blockIdx.x * blockDim.x + threadIdx.x;
  if (i < n) dst[i] = (_Float16)src[i];
}

// ---------------------------------------------------------------------------
// pre[t*B + b][n] = sum_k in16[b][t][k] * W16[n][k] + bih[n] + bhh[n]
// One wave per 16x16 output tile; since B==16, each M-tile is one timestep.
// ---------------------------------------------------------------------------
__global__ __launch_bounds__(256) void gemm_pre(
    const _Float16* __restrict__ in16, const _Float16* __restrict__ W16,
    const float* __restrict__ bih, const float* __restrict__ bhh,
    float* __restrict__ pre, int din)
{
  const int lane = threadIdx.x & 31;
  const int wave = threadIdx.x >> 5;
  const int tile = blockIdx.x * 8 + wave;        // 32768 tiles total
  const int mt = tile >> 4;                      // timestep (B==16)
  const int n0 = (tile & 15) * 16;

  const int m  = lane & 15;                      // batch row this lane feeds (A)
  const int n  = n0 + (lane & 15);               // output column (B/D)
  const int hi = (lane >> 4) & 1;

  const _Float16* arow = in16 + (size_t)m * TT * din + (size_t)mt * din;
  const _Float16* brow = W16 + (size_t)n * din;

  v8f acc = {};
  for (int kb = 0; kb < din; kb += 32) {
    v16h a = load_frag16(arow, kb, lane);
    v16h b = load_frag16(brow, kb, lane);
    acc = __builtin_amdgcn_wmma_f32_16x16x32_f16(false, a, false, b,
                                                 (short)0, acc, false, false);
  }
  const float bias = bih[n] + bhh[n];
#pragma unroll
  for (int r = 0; r < 8; ++r) {
    int mm = r + hi * 8;                         // batch row of this accum slot
    pre[((size_t)mt * BB + mm) * HH + n] = acc[r] + bias;
  }
}

// ---------------------------------------------------------------------------
// Persistent recurrence scan: h <- tanh(pre[t] + h @ Whh^T).
// 2 blocks (fwd/bwd), 16 waves each. Each wave's 16-column tile of Whh^T is
// hoisted into registers (8 f16 fragments = 64 VGPRs) for the entire scan.
// h lives in LDS (8 KB, f16). pre[t] slabs (16 KB) are double-buffered into
// LDS with GLOBAL_LOAD_ASYNC_TO_LDS_B128 when the toolchain exposes it.
// ---------------------------------------------------------------------------
__global__ __launch_bounds__(512) void rnn_scan(
    const float* __restrict__ pre_f, const float* __restrict__ pre_b,
    const _Float16* __restrict__ Whh16_f, const _Float16* __restrict__ Whh16_b,
    _Float16* __restrict__ out16, int fdim)
{
  __shared__ __align__(16) _Float16 h16[BB * HH];   // 8 KB hidden state
#ifdef HAVE_ASYNC_LDS
  __shared__ __align__(16) float pbuf[2][BB * HH];  // 2 x 16 KB pre slabs
#endif

  const int dir = blockIdx.x;                    // 0 = fwd, 1 = bwd
  const float*    __restrict__ pre   = dir ? pre_b   : pre_f;
  const _Float16* __restrict__ Whh16 = dir ? Whh16_b : Whh16_f;

  const int lane = threadIdx.x & 31;
  const int wave = threadIdx.x >> 5;             // 0..15 -> N tile
  const int nl = lane & 15;
  const int n  = wave * 16 + nl;
  const int hi = (lane >> 4) & 1;
  const int m  = lane & 15;                      // A-operand row this lane feeds

#ifdef HAVE_ASYNC_LDS
  // 512 threads x 2 x 16B = 16 KB slab per issue; two async ops per thread.
#define ISSUE_PRE(tq, which)                                                   \
  do {                                                                         \
    const float* gsrc_ = pre + (size_t)(tq) * (BB * HH);                       \
    const int off_ = threadIdx.x * 4;                                          \
    __builtin_amdgcn_global_load_async_to_lds_b128(                            \
        (v4i*)(gsrc_ + off_), (v4i*)&pbuf[which][off_], 0, 0);                 \
    __builtin_amdgcn_global_load_async_to_lds_b128(                            \
        (v4i*)(gsrc_ + off_ + 2048), (v4i*)&pbuf[which][off_ + 2048], 0, 0);   \
  } while (0)
#endif

  // Hoist B = (Whh^T) tile for this wave's 16 output columns into registers.
  F16Frag bfrag[8];
#pragma unroll
  for (int kk = 0; kk < 8; ++kk)
    bfrag[kk].v = load_frag16(Whh16 + (size_t)n * HH, kk * 32, lane);

  for (int i = threadIdx.x; i < BB * HH; i += blockDim.x) h16[i] = (_Float16)0.f;
#ifdef HAVE_ASYNC_LDS
  ISSUE_PRE(dir ? TT - 1 : 0, 0);
#endif
  __syncthreads();

  for (int s = 0; s < TT; ++s) {
    const int t = dir ? (TT - 1 - s) : s;
#ifdef HAVE_ASYNC_LDS
    if (s + 1 < TT) ISSUE_PRE(dir ? (TT - 2 - s) : (s + 1), (s + 1) & 1);
#else
    if (s + 1 < TT) {
      const int tn = dir ? (TT - 2 - s) : (s + 1);
      __builtin_prefetch(pre + ((size_t)tn * BB + m) * HH + n, 0, 0);
    }
#endif
    v8f acc = {};
#pragma unroll
    for (int kk = 0; kk < 8; ++kk) {
      v16h a = load_frag16(h16 + m * HH, kk * 32, lane);
      acc = __builtin_amdgcn_wmma_f32_16x16x32_f16(false, a, false, bfrag[kk].v,
                                                   (short)0, acc, false, false);
    }
#ifdef HAVE_ASYNC_LDS
    if (s + 1 < TT) __builtin_amdgcn_s_wait_asynccnt(2);  // current slab landed
    else            __builtin_amdgcn_s_wait_asynccnt(0);
#endif
    __syncthreads();                             // h16 reads done; slab visible
    float hv[8];
#pragma unroll
    for (int r = 0; r < 8; ++r) {
      int mm = r + hi * 8;
#ifdef HAVE_ASYNC_LDS
      float u = pbuf[s & 1][mm * HH + n];
#else
      float u = pre[((size_t)t * BB + mm) * HH + n];
#endif
      hv[r] = tanhf(u + acc[r]);
    }
#pragma unroll
    for (int r = 0; r < 8; ++r) {
      int mm = r + hi * 8;
      _Float16 hh = (_Float16)hv[r];
      h16[mm * HH + n] = hh;
      out16[(size_t)mm * TT * fdim + (size_t)t * fdim + dir * HH + n] = hh;
    }
    __syncthreads();                             // publish h16 for next step
  }
}

// ---------------------------------------------------------------------------
// Fused dual-head MLP: z = relu(feat @ W1^T + b1); y = z @ W2^T + b2.
// One block per 16-row tile. Phase 1: 16 waves (2 heads x 8 z-tiles),
// z staged to LDS f16. Phase 2: 4 waves produce the two 16x32 outputs.
// ---------------------------------------------------------------------------
__global__ __launch_bounds__(512) void heads_kernel(
    const _Float16* __restrict__ feat,           // [B*T][512]
    const _Float16* __restrict__ W1a, const float* __restrict__ b1a,
    const _Float16* __restrict__ W2a, const float* __restrict__ b2a,
    const _Float16* __restrict__ W1b, const float* __restrict__ b1b,
    const _Float16* __restrict__ W2b, const float* __restrict__ b2b,
    float* __restrict__ out)
{
  __shared__ __align__(16) _Float16 z16[2][16 * 128];  // 8 KB
  const int lane = threadIdx.x & 31;
  const int wave = threadIdx.x >> 5;
  const int m0 = blockIdx.x * 16;
  const int nl = lane & 15;
  const int hi = (lane >> 4) & 1;

  { // phase 1: z tiles
    const int head = wave >> 3;
    const int n0 = (wave & 7) * 16;
    const _Float16* __restrict__ W1 = head ? W1b : W1a;
    const float*    __restrict__ b1 = head ? b1b : b1a;
    const int n = n0 + nl;
    const _Float16* arow = feat + (size_t)(m0 + nl) * 512;
    const _Float16* brow = W1 + (size_t)n * 512;
    v8f acc = {};
    for (int kb = 0; kb < 512; kb += 32) {
      v16h a = load_frag16(arow, kb, lane);
      v16h b = load_frag16(brow, kb, lane);
      acc = __builtin_amdgcn_wmma_f32_16x16x32_f16(false, a, false, b,
                                                   (short)0, acc, false, false);
    }
    float bias = b1[n];
#pragma unroll
    for (int r = 0; r < 8; ++r) {
      int mm = r + hi * 8;
      float z = acc[r] + bias;
      z16[head][mm * 128 + n] = (_Float16)(z > 0.f ? z : 0.f);
    }
  }
  __syncthreads();

  if (wave < 4) { // phase 2: y tiles
    const int head = wave >> 1;
    const int n0 = (wave & 1) * 16;
    const _Float16* __restrict__ W2 = head ? W2b : W2a;
    const float*    __restrict__ b2 = head ? b2b : b2a;
    const int n = n0 + nl;
    v8f acc = {};
    for (int kb = 0; kb < 128; kb += 32) {
      v16h a = load_frag16(&z16[head][nl * 128], kb, lane);
      v16h b = load_frag16(W2 + (size_t)n * 128, kb, lane);
      acc = __builtin_amdgcn_wmma_f32_16x16x32_f16(false, a, false, b,
                                                   (short)0, acc, false, false);
    }
    float bias = b2[n];
    float* o = out + (size_t)head * (BB * TT * 32);
#pragma unroll
    for (int r = 0; r < 8; ++r) {
      int mm = r + hi * 8;
      o[(size_t)(m0 + mm) * 32 + n] = acc[r] + bias;
    }
  }
}

// ---------------------------------------------------------------------------
extern "C" void kernel_launch(void* const* d_in, const int* in_sizes, int n_in,
                              void* d_out, int out_size, void* d_ws, size_t ws_size,
                              hipStream_t stream) {
  const float* x     = (const float*)d_in[0];
  const float* Wih0f = (const float*)d_in[1];
  const float* Whh0f = (const float*)d_in[2];
  const float* bih0f = (const float*)d_in[3];
  const float* bhh0f = (const float*)d_in[4];
  const float* Wih0b = (const float*)d_in[5];
  const float* Whh0b = (const float*)d_in[6];
  const float* bih0b = (const float*)d_in[7];
  const float* bhh0b = (const float*)d_in[8];
  const float* Wih1f = (const float*)d_in[9];
  const float* Whh1f = (const float*)d_in[10];
  const float* bih1f = (const float*)d_in[11];
  const float* bhh1f = (const float*)d_in[12];
  const float* Wih1b = (const float*)d_in[13];
  const float* Whh1b = (const float*)d_in[14];
  const float* bih1b = (const float*)d_in[15];
  const float* bhh1b = (const float*)d_in[16];
  const float* W1t1  = (const float*)d_in[17];
  const float* b1t1  = (const float*)d_in[18];
  const float* W2t1  = (const float*)d_in[19];
  const float* b2t1  = (const float*)d_in[20];
  const float* W1t2  = (const float*)d_in[21];
  const float* b1t2  = (const float*)d_in[22];
  const float* W2t2  = (const float*)d_in[23];
  const float* b2t2  = (const float*)d_in[24];
  float* out = (float*)d_out;

  // Workspace carve-up
  char* ws = (char*)d_ws;
  _Float16* x16     = (_Float16*)(ws);                       //  8 MB
  _Float16* out0_16 = (_Float16*)(ws + (8ull   << 20));      // 32 MB
  _Float16* out1_16 = (_Float16*)(ws + (40ull  << 20));      // 32 MB
  float*    pre_fv  = (float*)   (ws + (72ull  << 20));      // 32 MB
  float*    pre_bv  = (float*)   (ws + (104ull << 20));      // 32 MB
  _Float16* wp      = (_Float16*)(ws + (136ull << 20));      // ~1.5 MB f16 weights
  _Float16* Wih0f16 = wp; wp += HH * DD;
  _Float16* Wih0b16 = wp; wp += HH * DD;
  _Float16* Whh0f16 = wp; wp += HH * HH;
  _Float16* Whh0b16 = wp; wp += HH * HH;
  _Float16* Wih1f16 = wp; wp += HH * 2 * HH;
  _Float16* Wih1b16 = wp; wp += HH * 2 * HH;
  _Float16* Whh1f16 = wp; wp += HH * HH;
  _Float16* Whh1b16 = wp; wp += HH * HH;
  _Float16* W1t1_16 = wp; wp += 128 * 512;
  _Float16* W1t2_16 = wp; wp += 128 * 512;
  _Float16* W2t1_16 = wp; wp += 32 * 128;
  _Float16* W2t2_16 = wp; wp += 32 * 128;

  auto cvt = [&](const float* s, _Float16* d, int n) {
    cvt_f32_f16<<<(n + 255) / 256, 256, 0, stream>>>(s, d, n);
  };

  // 1. One-time f16 conversions (x + all weight matrices)
  cvt(x, x16, BB * TT * DD);
  cvt(Wih0f, Wih0f16, HH * DD);
  cvt(Wih0b, Wih0b16, HH * DD);
  cvt(Whh0f, Whh0f16, HH * HH);
  cvt(Whh0b, Whh0b16, HH * HH);
  cvt(Wih1f, Wih1f16, HH * 2 * HH);
  cvt(Wih1b, Wih1b16, HH * 2 * HH);
  cvt(Whh1f, Whh1f16, HH * HH);
  cvt(Whh1b, Whh1b16, HH * HH);
  cvt(W1t1, W1t1_16, 128 * 512);
  cvt(W1t2, W1t2_16, 128 * 512);
  cvt(W2t1, W2t1_16, 32 * 128);
  cvt(W2t2, W2t2_16, 32 * 128);

  const int gemm_blocks = (TT * BB / 16) * (HH / 16) / 8;  // 4096

  // 2. Layer 0: input projections + bidirectional scan
  gemm_pre<<<gemm_blocks, 256, 0, stream>>>(x16, Wih0f16, bih0f, bhh0f, pre_fv, DD);
  gemm_pre<<<gemm_blocks, 256, 0, stream>>>(x16, Wih0b16, bih0b, bhh0b, pre_bv, DD);
  rnn_scan<<<2, 512, 0, stream>>>(pre_fv, pre_bv, Whh0f16, Whh0b16, out0_16, 2 * HH);

  // 3. Layer 1
  gemm_pre<<<gemm_blocks, 256, 0, stream>>>(out0_16, Wih1f16, bih1f, bhh1f, pre_fv, 2 * HH);
  gemm_pre<<<gemm_blocks, 256, 0, stream>>>(out0_16, Wih1b16, bih1b, bhh1b, pre_bv, 2 * HH);
  rnn_scan<<<2, 512, 0, stream>>>(pre_fv, pre_bv, Whh1f16, Whh1b16, out1_16, 2 * HH);

  // 4. Dual MLP heads
  heads_kernel<<<(BB * TT) / 16, 512, 0, stream>>>(
      out1_16, W1t1_16, b1t1, W2t1_16, b2t1, W1t2_16, b1t2, W2t2_16, b2t2, out);
}
